// BiLSTMCRF_22608707846774
// MI455X (gfx1250) — matrix operand order repaired and checked by reading, not compiled, not run on hardware
//
#include <hip/hip_runtime.h>
#include <hip/hip_bf16.h>
#include <math.h>
#include <stdint.h>

typedef __bf16 bf16;
typedef __attribute__((ext_vector_type(4)))  __bf16 v4bf;
typedef __attribute__((ext_vector_type(8)))  __bf16 v8bf;
typedef __attribute__((ext_vector_type(16))) __bf16 v16bf;
typedef __attribute__((ext_vector_type(8)))  float  v8f;
typedef __attribute__((ext_vector_type(4)))  int    v4i;

#define BB 64
#define SS 512
#define EE 256
#define HH 512
#define TT 9
#define G4H 2048            // 4*H
#define KC 768              // E + H  (combined K for fused step GEMM)
#define AP 776              // LDS A row pitch (bf16 elems), 16B-aligned stride

#if __has_builtin(__builtin_amdgcn_global_load_async_to_lds_b128)
#define HAVE_ASYNC_LDS 1
#else
#define HAVE_ASYNC_LDS 0
#endif

#define GAS __attribute__((address_space(1)))
#define LAS __attribute__((address_space(3)))

// ---------------- workspace layout (bytes) ----------------
#define OFF_X   ((size_t)0)                       // x bf16 [B][S][E]            16.78 MB
#define OFF_WC  (OFF_X  + (size_t)BB*SS*EE*2)     // Wc bf16 [2][2048][768]       6.29 MB
#define OFF_H   (OFF_WC + (size_t)2*G4H*KC*2)     // h bf16 [2][B][S][H]         67.11 MB
#define OFF_W2  (OFF_H  + (size_t)2*BB*SS*HH*2)   // Wt2 bf16 [16][1024]
#define OFF_B2  (OFF_W2 + (size_t)16*1024*2)      // bc2 f32 [16]
#define OFF_BP  (OFF_B2 + (size_t)64)             // viterbi bp u8 [B][S][16]

// ---------------- prep: combined recurrent weight, bf16, transposed ----------
// Wc[d][n][k] = k<E ? W_d[k][n] : U_d[k-E][n]
__global__ void prep_wc(const float* __restrict__ Wf, const float* __restrict__ Uf,
                        const float* __restrict__ Wb, const float* __restrict__ Ub,
                        bf16* __restrict__ wc) {
  const int d = blockIdx.x >> 11;
  const int n = blockIdx.x & 2047;
  const float* W = d ? Wb : Wf;
  const float* U = d ? Ub : Uf;
  bf16* dst = wc + ((size_t)d * G4H + n) * KC;
  for (int k = threadIdx.x; k < KC; k += 256) {
    float v = (k < EE) ? W[(size_t)k * G4H + n] : U[(size_t)(k - EE) * G4H + n];
    dst[k] = (bf16)v;
  }
}

// ---------------- prep: fold Dense+CRF:  Wt2[t][k] = sum_j Wd[k][j]*crfW[j][t]
__global__ void prep_w2(const float* __restrict__ Wd_, const float* __restrict__ bd_,
                        const float* __restrict__ crfW, const float* __restrict__ crfb,
                        bf16* __restrict__ wt2, float* __restrict__ bc2) {
  for (int idx = threadIdx.x; idx < 16 * 1024; idx += 256) {
    const int t = idx >> 10, k = idx & 1023;
    float v = 0.f;
    if (t < TT)
      for (int j = 0; j < TT; ++j) v += Wd_[(size_t)k * TT + j] * crfW[j * TT + t];
    wt2[idx] = (bf16)v;
  }
  if (threadIdx.x < 16) {
    const int t = threadIdx.x;
    float v = 0.f;
    if (t < TT) {
      for (int j = 0; j < TT; ++j) v += bd_[j] * crfW[j * TT + t];
      v += crfb[t];
    }
    bc2[t] = v;
  }
}

// ---------------- prep: embedding gather + f32->bf16 cast ---------------------
__global__ void embed_cast(const int* __restrict__ inputs, const float* __restrict__ emb,
                           bf16* __restrict__ x16) {
  const size_t i = ((size_t)blockIdx.x * 256 + threadIdx.x) * 4;
  const size_t r = i >> 8;             // row in [B*S)
  const int    e = (int)(i & 255);
  const float4 f = *(const float4*)(emb + (size_t)inputs[r] * EE + e);
  v4bf o; o[0] = (bf16)f.x; o[1] = (bf16)f.y; o[2] = (bf16)f.z; o[3] = (bf16)f.w;
  *(v4bf*)(x16 + i) = o;
}

// ---------------- persistent fused LSTM (one WG = 16 batch rows, one dir) -----
// grid (4,2): x = batch group, y = direction. Double-buffered LDS A = [x_t | h].
__global__ __launch_bounds__(256) void lstm_kernel(
    const bf16* __restrict__ xall, const bf16* __restrict__ wc_all,
    const float* __restrict__ b_fw, const float* __restrict__ b_bw,
    bf16* __restrict__ h_out) {
  __shared__ bf16 A[2][16][AP];

  const int bg   = blockIdx.x;                 // 0..3
  const int d    = blockIdx.y;                 // 0..1
  const int tid  = threadIdx.x;
  const int lane = tid & 31;
  const int wave = tid >> 5;                   // 8 waves
  const int lr   = lane & 15;
  const int lh   = (lane >> 4) & 1;
  const float* bias = d ? b_bw : b_fw;

  // weight base in EXPLICIT AS1 (so the opaque asm below can't demote loads to
  // FLAT); made loop-variant each step so the compiler cannot hoist the weight
  // fragment loads out of the t-loop (=> scratch spills).
  const GAS bf16* wbase =
      (const GAS bf16*)(uintptr_t)(wc_all + (size_t)d * G4H * KC);

  // zero h-region of buffer 0 (initial hidden state)
  for (int i = tid; i < 16 * (AP - EE); i += 256) {
    const int r = i / (AP - EE), c = i % (AP - EE);
    A[0][r][EE + c] = (bf16)0.f;
  }
  // stage x_0 into buffer 0 (synchronous)
  {
    const int p0 = d ? (SS - 1) : 0;
    const int r = tid >> 4, ch = tid & 15;
    const v8bf* src = (const v8bf*)(xall + ((size_t)(bg * 16 + r) * SS + p0) * EE + ch * 16);
    *(v8bf*)&A[0][r][ch * 16]     = src[0];
    *(v8bf*)&A[0][r][ch * 16 + 8] = src[1];
  }
  __syncthreads();

  const int hbase = wave * 64;                 // this wave owns h cols [hbase, hbase+64)
  v8f cstate[4];
  #pragma unroll
  for (int s4 = 0; s4 < 4; ++s4)
    #pragma unroll
    for (int e = 0; e < 8; ++e) cstate[s4][e] = 0.f;

  for (int t = 0; t < SS; ++t) {
    asm volatile("" : "+s"(wbase));            // block cross-iteration hoisting
    const int cur = t & 1, nxt = cur ^ 1;
    const int p = d ? (SS - 1 - t) : t;

    // stage x_{t+1} into the other buffer (disjoint from all cur readers)
    if (t < SS - 1) {
      const int pn = d ? (SS - 2 - t) : (t + 1);
      const int r = tid >> 4, ch = tid & 15;
      const bf16* gsrc = xall + ((size_t)(bg * 16 + r) * SS + pn) * EE + ch * 16;
      bf16* ldst = &A[nxt][r][ch * 16];
#if HAVE_ASYNC_LDS
      __builtin_amdgcn_global_load_async_to_lds_b128(
          (GAS v4i*)(uintptr_t)gsrc,
          (LAS v4i*)ldst, 0, 0);
      __builtin_amdgcn_global_load_async_to_lds_b128(
          (GAS v4i*)(uintptr_t)(gsrc + 8),
          (LAS v4i*)(ldst + 8), 0, 0);
#else
      const v8bf* src = (const v8bf*)gsrc;
      *(v8bf*)ldst       = src[0];
      *(v8bf*)(ldst + 8) = src[1];
#endif
    }

    #pragma unroll
    for (int sub = 0; sub < 4; ++sub) {
      const int hc0 = hbase + sub * 16;        // h column base of this tile group
      v8f acc[4];                               // i,f,g,o tiles for same h cols
      #pragma unroll
      for (int g4 = 0; g4 < 4; ++g4) {
        const float bv = bias[g4 * HH + hc0 + lr];
        #pragma unroll
        for (int e = 0; e < 8; ++e) acc[g4][e] = bv;
      }

      #pragma unroll 4
      for (int kt = 0; kt < KC / 32; ++kt) {
        // A fragment: row=lr, contiguous K16 {klo..klo+15} (matches B)
        const int klo = kt * 32 + lh * 16;
        const v8bf a0 = *(const v8bf*)&A[cur][lr][klo];
        const v8bf a1 = *(const v8bf*)&A[cur][lr][klo + 8];
        const v16bf af = __builtin_shufflevector(a0, a1,
            0, 1, 2, 3, 4, 5, 6, 7, 8, 9, 10, 11, 12, 13, 14, 15);
        const int kb = kt * 32 + lh * 16;      // B fragment: contiguous K16
        #pragma unroll
        for (int g4 = 0; g4 < 4; ++g4) {
          const int n = g4 * HH + hc0 + lr;
          const v16bf bfrag = *(const GAS v16bf*)(wbase + (size_t)n * KC + kb);
          acc[g4] = __builtin_amdgcn_wmma_f32_16x16x32_bf16(
              false, af, false, bfrag, (short)0, acc[g4], false, false);
        }
      }

      // gate math; C/D layout aligns i/f/g/o elementwise (col=lr, row=e+8*lh)
      v8f hv;
      #pragma unroll
      for (int e = 0; e < 8; ++e) {
        const float zi = acc[0][e], zf = acc[1][e], zg = acc[2][e], zo = acc[3][e];
        const float si = 1.f / (1.f + __expf(-zi));
        const float sf = 1.f / (1.f + __expf(-zf));
        const float so = 1.f / (1.f + __expf(-zo));
        const float cc = sf * cstate[sub][e] + si * tanhf(zg);
        cstate[sub][e] = cc;
        hv[e] = so * tanhf(cc);
      }
      #pragma unroll
      for (int e = 0; e < 8; ++e) {
        const int row = e + lh * 8;
        const int col = hc0 + lr;
        const bf16 hb = (bf16)hv[e];
        A[nxt][row][EE + col] = hb;            // feed next step
        h_out[(((size_t)(d * BB + bg * 16 + row)) * SS + p) * HH + col] = hb;
      }
    }

#if HAVE_ASYNC_LDS
#if __has_builtin(__builtin_amdgcn_s_wait_asynccnt)
    __builtin_amdgcn_s_wait_asynccnt(0);
#else
    asm volatile("s_wait_asynccnt 0" ::: "memory");
#endif
#endif
    __syncthreads();                            // next-step A fully staged
  }
}

// ---------------- potentials: [B*S,1024] @ [1024,16(9)] with WMMA -------------
__global__ __launch_bounds__(256) void pots_kernel(
    const bf16* __restrict__ h_ws, const bf16* __restrict__ wt2,
    const float* __restrict__ bc2, const float* __restrict__ left_b,
    const float* __restrict__ right_b, float* __restrict__ pots) {
  const int tid = threadIdx.x, lane = tid & 31, wave = tid >> 5;
  const int lr = lane & 15, lh = (lane >> 4) & 1;
  const int mtile = blockIdx.x * 8 + wave;     // 0..2047
  const int bs_a = mtile * 16 + lr;            // A-matrix row this lane loads

  v8f acc;
  const float bv = bc2[lr];
  #pragma unroll
  for (int e = 0; e < 8; ++e) acc[e] = bv;

  for (int kt = 0; kt < 32; ++kt) {
    const int klo  = kt * 32 + lh * 16;
    const int dsel = klo >> 9;                 // 0 = fw half, 1 = bw half
    const int k0   = klo & 511;
    const bf16* hrow = h_ws +
        (((size_t)dsel * BB + (bs_a >> 9)) * SS + (bs_a & 511)) * HH;
    const v8bf a0 = *(const v8bf*)(hrow + k0);
    const v8bf a1 = *(const v8bf*)(hrow + k0 + 8);
    const v16bf af = __builtin_shufflevector(a0, a1,
        0, 1, 2, 3, 4, 5, 6, 7, 8, 9, 10, 11, 12, 13, 14, 15);
    const int kb = kt * 32 + lh * 16;
    const v16bf bfrag = *(const v16bf*)(wt2 + (size_t)lr * 1024 + kb);
    acc = __builtin_amdgcn_wmma_f32_16x16x32_bf16(
        false, af, false, bfrag, (short)0, acc, false, false);
  }

  const int t = lr;
  if (t < TT) {
    const float lb = left_b[t], rb = right_b[t];
    #pragma unroll
    for (int e = 0; e < 8; ++e) {
      const int m = e + lh * 8;
      const int bs2 = mtile * 16 + m;
      const int s = bs2 & 511;
      float v = acc[e];
      if (s == 0)      v += lb;
      if (s == SS - 1) v += rb;
      pots[(size_t)bs2 * TT + t] = v;
    }
  }
}

// ---------------- viterbi: one lane per batch element -------------------------
__global__ void viterbi_kernel(const float* __restrict__ pots,
                               const float* __restrict__ trans,
                               unsigned char* __restrict__ bp,
                               float* __restrict__ dec) {
  __shared__ float tr[TT * TT];
  const int tid = threadIdx.x;
  for (int i = tid; i < TT * TT; i += 64) tr[i] = trans[i];
  __syncthreads();

  const int b = tid;                           // 0..63
  const float* pb = pots + (size_t)b * SS * TT;
  float alpha[TT];
  #pragma unroll
  for (int j = 0; j < TT; ++j) alpha[j] = pb[j];

  for (int s = 1; s < SS; ++s) {
    float na[TT];
    #pragma unroll
    for (int j = 0; j < TT; ++j) {
      float best = alpha[0] + tr[j];
      int bi = 0;
      #pragma unroll
      for (int i = 1; i < TT; ++i) {
        const float v = alpha[i] + tr[i * TT + j];
        if (v > best) { best = v; bi = i; }
      }
      bp[((size_t)b * SS + s) * 16 + j] = (unsigned char)bi;
      na[j] = best + pb[(size_t)s * TT + j];
    }
    #pragma unroll
    for (int j = 0; j < TT; ++j) alpha[j] = na[j];
  }

  int last = 0; float bestv = alpha[0];
  #pragma unroll
  for (int j = 1; j < TT; ++j)
    if (alpha[j] > bestv) { bestv = alpha[j]; last = j; }

  dec[(size_t)b * SS + SS - 1] = (float)last;
  int cur = last;
  for (int s = SS - 1; s >= 1; --s) {
    cur = bp[((size_t)b * SS + s) * 16 + cur];
    dec[(size_t)b * SS + s - 1] = (float)cur;
  }
}

// ---------------- launcher ----------------------------------------------------
extern "C" void kernel_launch(void* const* d_in, const int* in_sizes, int n_in,
                              void* d_out, int out_size, void* d_ws, size_t ws_size,
                              hipStream_t stream) {
  (void)in_sizes; (void)n_in; (void)out_size; (void)ws_size;
  const int*   inputs  = (const int*)  d_in[0];
  const float* emb     = (const float*)d_in[1];
  const float* W_fw    = (const float*)d_in[2];
  const float* U_fw    = (const float*)d_in[3];
  const float* b_fw    = (const float*)d_in[4];
  const float* W_bw    = (const float*)d_in[5];
  const float* U_bw    = (const float*)d_in[6];
  const float* b_bw    = (const float*)d_in[7];
  const float* Wd_     = (const float*)d_in[8];
  const float* bd_     = (const float*)d_in[9];
  const float* crf_W   = (const float*)d_in[10];
  const float* crf_b   = (const float*)d_in[11];
  const float* trans   = (const float*)d_in[12];
  const float* left_b  = (const float*)d_in[13];
  const float* right_b = (const float*)d_in[14];

  char* ws = (char*)d_ws;
  bf16*  x16 = (bf16*)(ws + OFF_X);
  bf16*  wc  = (bf16*)(ws + OFF_WC);
  bf16*  hws = (bf16*)(ws + OFF_H);
  bf16*  wt2 = (bf16*)(ws + OFF_W2);
  float* bc2 = (float*)(ws + OFF_B2);
  unsigned char* bp = (unsigned char*)(ws + OFF_BP);

  float* out  = (float*)d_out;                 // decoded [B,S] first
  float* pots = out + (size_t)BB * SS;         // then pots [B,S,T]

  prep_wc  <<<2 * G4H, 256, 0, stream>>>(W_fw, U_fw, W_bw, U_bw, wc);
  prep_w2  <<<1, 256, 0, stream>>>(Wd_, bd_, crf_W, crf_b, wt2, bc2);
  embed_cast<<<(BB * SS * EE) / (256 * 4), 256, 0, stream>>>(inputs, emb, x16);

  dim3 g(4, 2);
  lstm_kernel<<<g, 256, 0, stream>>>(x16, wc, b_fw, b_bw, hws);

  pots_kernel<<<(BB * SS / 16) / 8, 256, 0, stream>>>(hws, wt2, bc2,
                                                      left_b, right_b, pots);
  viterbi_kernel<<<1, 64, 0, stream>>>(pots, trans, bp, out);
}
